// LongFlowNetCorr_41970420417269
// MI455X (gfx1250) — matrix operand order repaired and checked by reading, not compiled, not run on hardware
//
#include <hip/hip_runtime.h>
#include <hip/hip_bf16.h>
#include <stdint.h>

typedef __bf16 bf16;
typedef __attribute__((ext_vector_type(16))) __bf16 v16bf;
typedef __attribute__((ext_vector_type(8)))  float  v8f;

#define BN 8
#define HH 128
#define WW 128
#define NPIX ((size_t)BN * HH * WW)   // 131072 pixels

__device__ __forceinline__ float bflo(unsigned u) { return __uint_as_float(u << 16); }
__device__ __forceinline__ float bfhi(unsigned u) { return __uint_as_float(u & 0xffff0000u); }

// ---------------------------------------------------------------------------
// CDNA5 async global->LDS copy (ASYNCcnt path), builtin if present else asm.
// LDS byte address = low 32 bits of the generic pointer.
// ---------------------------------------------------------------------------
__device__ __forceinline__ void async_copy_b32(const unsigned* g, unsigned* l) {
#if __has_builtin(__builtin_amdgcn_global_load_async_to_lds_b32)
    __builtin_amdgcn_global_load_async_to_lds_b32(
        (__attribute__((address_space(1))) int*)(uintptr_t)g,
        (__attribute__((address_space(3))) int*)(unsigned)(uintptr_t)l, 0, 0);
#else
    unsigned lba = (unsigned)(uintptr_t)l;
    asm volatile("global_load_async_to_lds_b32 %0, %1, off"
                 :: "v"(lba), "v"(g) : "memory");
#endif
}

__device__ __forceinline__ void wait_async0() {
#if __has_builtin(__builtin_amdgcn_s_wait_asynccnt)
    __builtin_amdgcn_s_wait_asynccnt(0);
#else
    asm volatile("s_wait_asynccnt 0" ::: "memory");
#endif
}

// ---------------------------------------------------------------------------
// NCHW f32 -> NHWC bf16 transpose (C = 128), LDS-staged so both sides coalesce
// ---------------------------------------------------------------------------
__global__ void k_nchw_to_nhwc(const float* __restrict__ in, bf16* __restrict__ out) {
    __shared__ bf16 t[32][130];                 // +2 pad vs bank conflicts
    int id = blockIdx.x;                        // 4 x-tiles * 128 y * 8 b
    int xt = id & 3, y = (id >> 2) & 127, b = id >> 9;
    int x0 = xt * 32;
    int tid = threadIdx.x, xi = tid & 31;
    for (int c = tid >> 5; c < 128; c += 8) {   // coalesced over x per fixed c
        float v = in[(((size_t)b * 128 + c) * HH + y) * WW + x0 + xi];
        t[xi][c] = (bf16)v;
    }
    __syncthreads();
    unsigned* o32 = (unsigned*)out;
    size_t pixbase = ((size_t)b * HH + y) * WW + x0;
    for (int i = tid; i < 32 * 64; i += 256) {  // coalesced over channel-pairs
        int px = i >> 6, c2 = i & 63;
        union { bf16 h[2]; unsigned u; } pk;
        pk.h[0] = t[px][c2 * 2];
        pk.h[1] = t[px][c2 * 2 + 1];
        o32[(pixbase + px) * 64 + c2] = pk.u;
    }
}

// ---------------------------------------------------------------------------
// Correlation (SR=4, 81 offsets, mean over 128 ch) + concat into 224-ch act0:
//   ch 0..127 = x1, ch 128..208 = corr, ch 209..223 = zero (K padding)
// One wave per pixel; x1 vector staged in LDS, lanes sweep offsets.
// ---------------------------------------------------------------------------
__global__ void k_corr_concat(const bf16* __restrict__ xw1, const bf16* __restrict__ xw2,
                              bf16* __restrict__ act0) {
    __shared__ unsigned x1s[8][64];
    int wave = threadIdx.x >> 5, lane = threadIdx.x & 31;
    size_t p = (size_t)blockIdx.x * 8 + wave;
    int x = (int)(p % WW);
    int y = (int)((p / WW) % HH);
    int b = (int)(p / ((size_t)WW * HH));

    const unsigned* v1 = (const unsigned*)xw1 + p * 64;
    unsigned u0 = v1[lane], u1 = v1[32 + lane];
    x1s[wave][lane] = u0;
    x1s[wave][32 + lane] = u1;

    unsigned* a32 = (unsigned*)act0;            // copy x1 channels (0..127)
    a32[p * 112 + lane] = u0;
    a32[p * 112 + 32 + lane] = u1;
    if (lane < 15) act0[p * 224 + 209 + lane] = (bf16)0.f;   // zero K-pad
    __syncthreads();

    for (int o = lane; o < 81; o += 32) {
        int dy = o / 9 - 4, dx = o % 9 - 4;
        int gy = y + dy, gx = x + dx;
        float acc = 0.f;
        if (gy >= 0 && gy < HH && gx >= 0 && gx < WW) {
            const unsigned* v2 = (const unsigned*)xw2 + ((((size_t)b * HH + gy) * WW + gx) << 6);
            #pragma unroll 8
            for (int c2 = 0; c2 < 64; ++c2) {
                unsigned a = x1s[wave][c2], bb = v2[c2];
                acc += bflo(a) * bflo(bb) + bfhi(a) * bfhi(bb);
            }
        }
        act0[p * 224 + 128 + o] = (bf16)(acc * (1.f / 128.f));
    }
}

// ---------------------------------------------------------------------------
// Weight swizzle: f32 [O][I][3][3] -> bf16 fragments in exact CDNA5 B-layout.
// Fragment f = (tap*KC + kc)*Ntiles + nt holds 32 lanes x 16 bf16 contiguous:
//   element(lane, j): K = kc*32 + 8*(lane>=16) + (j&7) + 16*(j>=8), N = nt*16 + (lane&15)
// Zero-fills K >= Cin and N >= Cout (K padding).
// ---------------------------------------------------------------------------
__global__ void k_swizzle_w(const float* __restrict__ w, bf16* __restrict__ out,
                            int Cin, int Cout, int Cpad, int Ntiles, int total) {
    int KC = Cpad >> 5;
    for (int idx = blockIdx.x * 256 + threadIdx.x; idx < total; idx += gridDim.x * 256) {
        int j = idx & 15;
        int lane = (idx >> 4) & 31;
        int frag = idx >> 9;
        int nt = frag % Ntiles;
        int kc = (frag / Ntiles) % KC;
        int tap = frag / (Ntiles * KC);
        int k = kc * 32 + ((lane >> 4) << 3) + (j & 7) + ((j >> 3) << 4);
        int n = nt * 16 + (lane & 15);
        float v = 0.f;
        if (n < Cout && k < Cin) v = w[((size_t)n * Cin + k) * 9 + tap];
        out[idx] = (bf16)v;
    }
}

// zero-fill context-input channels 34..63 (uints 17..31 of 32-uint rows)
__global__ void k_ctx_zero(unsigned* __restrict__ ctx0) {
    size_t total = NPIX * 15;
    for (size_t i = (size_t)blockIdx.x * 256 + threadIdx.x; i < total;
         i += (size_t)gridDim.x * 256) {
        size_t pp = i / 15;
        int j = (int)(i % 15);
        ctx0[pp * 32 + 17 + j] = 0u;
    }
}

// ---------------------------------------------------------------------------
// Implicit-GEMM 3x3 conv, bf16 WMMA, NHWC activations.
// Block = 256 thr (8 waves) = one M-tile of 16 pixels of one image row.
// Wave nt computes output channels [nt*16, nt*16+16).
// LDS: 3 dilated input rows x (16 + 2*dil) pixels x Cpad bf16, zero halo,
// filled with GLOBAL_LOAD_ASYNC_TO_LDS_B32 (ASYNCcnt) where in-range.
// K loop: 9 taps x Cpad/32, one v_wmma_f32_16x16x32_bf16 per step.
// ---------------------------------------------------------------------------
__global__ void k_conv3x3_wmma(const bf16* __restrict__ in, int in_stride, int Cpad,
                               const bf16* __restrict__ wsw, const float* __restrict__ bias,
                               bf16* __restrict__ out, int out_stride, int out_choff,
                               int Cout, int dil, int leaky, float* __restrict__ flo) {
    extern __shared__ __align__(16) char smem_raw[];
    bf16* lds = (bf16*)smem_raw;
    const int PXW = 16 + 2 * dil;

    int tile = blockIdx.x;                     // 8 x-tiles * 128 y * 8 b
    int tx = tile & 7, y = (tile >> 3) & 127, b = tile >> 10;
    int x0 = tx << 4;

    // ---- async LDS fill (2 channels per u32); zero halo via ds_store ----
    int Cp2 = Cpad >> 1;
    int n2 = 3 * PXW * Cp2;
    unsigned* lds32 = (unsigned*)lds;
    const unsigned* in32 = (const unsigned*)in;
    int ins2 = in_stride >> 1;
    for (int i = threadIdx.x; i < n2; i += 256) {
        int c2 = i % Cp2;
        int px = (i / Cp2) % PXW;
        int r = i / (Cp2 * PXW);
        int gy = y + (r - 1) * dil;
        int gx = x0 - dil + px;
        if (gy >= 0 && gy < HH && gx >= 0 && gx < WW) {
            const unsigned* gp = in32 + (((size_t)b * HH + gy) * WW + gx) * ins2 + c2;
            async_copy_b32(gp, lds32 + i);
        } else {
            lds32[i] = 0u;
        }
    }
    wait_async0();
    __syncthreads();

    int wave = threadIdx.x >> 5, lane = threadIdx.x & 31;
    int Ntiles = (Cout + 15) >> 4;
    if (wave >= Ntiles) return;                // wave-uniform; EXEC stays all-1s

    v8f acc = {0.f, 0.f, 0.f, 0.f, 0.f, 0.f, 0.f, 0.f};
    const int mrow = lane & 15;
    const int ahalf = (lane >> 4) << 3;        // A layout: K 0-7/16-23 vs 8-15/24-31
    const int KC = Cpad >> 5;
    const bf16* wlane = wsw + (size_t)lane * 16;

    for (int tap = 0; tap < 9; ++tap) {
        int ky = tap / 3, kx = tap % 3;
        const bf16* arow = lds + (size_t)(ky * PXW + mrow + kx * dil) * Cpad + ahalf;
        const bf16* wt = wlane + ((size_t)tap * KC * Ntiles + wave) * 512;
        for (int kc = 0; kc < Cpad; kc += 32) {
            union { v16bf v; uint4 q[2]; } A, Bf;
            A.q[0] = *(const uint4*)(arow + kc);        // ds_load_b128
            A.q[1] = *(const uint4*)(arow + kc + 16);   // ds_load_b128
            Bf.q[0] = *(const uint4*)(wt);              // global_load_b128
            Bf.q[1] = *(const uint4*)(wt + 8);
            __builtin_prefetch(wt + (size_t)Ntiles * 512, 0, 1);  // next B frag
            acc = __builtin_amdgcn_wmma_f32_16x16x32_bf16(
                false, A.v, false, Bf.v, (short)0, acc, false, false);
            wt += (size_t)Ntiles * 512;
        }
    }

    // ---- epilogue: C/D layout M = r + 8*(lane>=16), N = lane&15 ----
    int n = (wave << 4) + (lane & 15);
    int mofs = (lane >> 4) << 3;
    size_t pixbase = ((size_t)b * HH + y) * WW + x0;
    if (n < Cout) {
        float bv = bias[n];
        #pragma unroll
        for (int r = 0; r < 8; ++r) {
            int m = r + mofs;
            float v = acc[r] + bv;
            if (leaky) v = (v > 0.f) ? v : 0.1f * v;
            out[(pixbase + m) * out_stride + out_choff + n] = (bf16)v;
            if (flo) flo[(((size_t)b * 2 + n) * HH + y) * WW + x0 + m] = v;  // NCHW f32 head
        }
    }
}

// out[b,c,y,x] = flo_coarse + leaky(ctx_out)   (both already computed)
__global__ void k_final_add(const float* __restrict__ flo, const bf16* __restrict__ ypad,
                            float* __restrict__ out) {
    size_t hw = (size_t)HH * WW;
    size_t total = NPIX * 2;
    for (size_t i = (size_t)blockIdx.x * 256 + threadIdx.x; i < total;
         i += (size_t)gridDim.x * 256) {
        size_t b = i / (2 * hw);
        size_t c = (i / hw) & 1;
        size_t rem = i % hw;
        out[i] = flo[i] + (float)ypad[(b * hw + rem) * 16 + c];
    }
}

// ---------------------------------------------------------------------------
extern "C" void kernel_launch(void* const* d_in, const int* in_sizes, int n_in,
                              void* d_out, int out_size, void* d_ws, size_t ws_size,
                              hipStream_t stream) {
    (void)in_sizes; (void)n_in; (void)out_size; (void)ws_size;
    const float* x1 = (const float*)d_in[0];
    const float* x2 = (const float*)d_in[1];

    char* ws = (char*)d_ws;
    size_t off = 0;
    auto take = [&](size_t bytes) -> char* {
        char* p = ws + off;
        off += (bytes + 255) & ~(size_t)255;
        return p;
    };

    bf16* xw1 = (bf16*)take(NPIX * 128 * 2);   // x1 NHWC bf16, reused as bufA
    bf16* xw2 = (bf16*)take(NPIX * 128 * 2);   // x2 NHWC bf16, reused as bufB
    char* actR = take(NPIX * 224 * 2);         // act0; reused as ctx0/ypad/flo
    bf16* act0 = (bf16*)actR;
    bf16* ctx0 = (bf16*)actR;                                    // 64-ch ctx input
    bf16* ypad = (bf16*)(actR + NPIX * 64 * 2);                  // 16-ch ctx output
    float* flo = (float*)(actR + NPIX * 64 * 2 + NPIX * 16 * 2); // f32 NCHW flow
    bf16* bufA = xw1;
    bf16* bufB = xw2;

    struct LW { int widx, bidx, Cin, Cout, Cpad; };
    const LW L[13] = {
        { 2,  3, 209, 128, 224}, { 4,  5, 128, 128, 128}, { 6,  7, 128,  96, 128},
        { 8,  9,  96,  64,  96}, {10, 11,  64,  32,  64}, {12, 13,  32,   2,  32},
        {14, 15,  34, 128,  64}, {16, 17, 128, 128, 128}, {18, 19, 128, 128, 128},
        {20, 21, 128,  96, 128}, {22, 23,  96,  64,  96}, {24, 25,  64,  32,  64},
        {26, 27,  32,   2,  32}
    };
    bf16* wsw[13];
    for (int i = 0; i < 13; ++i) {
        int Nt = (L[i].Cout + 15) / 16;
        size_t elems = (size_t)9 * (L[i].Cpad / 32) * Nt * 512;
        wsw[i] = (bf16*)take(elems * 2);
    }

    // weight swizzle (small) + input transposes + correlation
    for (int i = 0; i < 13; ++i) {
        int Nt = (L[i].Cout + 15) / 16;
        int total = 9 * (L[i].Cpad / 32) * Nt * 512;
        k_swizzle_w<<<(total + 255) / 256, 256, 0, stream>>>(
            (const float*)d_in[L[i].widx], wsw[i], L[i].Cin, L[i].Cout, L[i].Cpad, Nt, total);
    }
    k_nchw_to_nhwc<<<4096, 256, 0, stream>>>(x1, xw1);
    k_nchw_to_nhwc<<<4096, 256, 0, stream>>>(x2, xw2);
    k_corr_concat<<<(unsigned)(NPIX / 8), 256, 0, stream>>>(xw1, xw2, act0);

    auto conv = [&](const bf16* in, int in_stride, int Cpad, int li,
                    bf16* outp, int out_stride, int choff, int dil, int leaky, float* flo_p) {
        size_t smem = (size_t)3 * (16 + 2 * dil) * Cpad * 2;
        k_conv3x3_wmma<<<8192, 256, smem, stream>>>(
            in, in_stride, Cpad, wsw[li], (const float*)d_in[L[li].bidx],
            outp, out_stride, choff, L[li].Cout, dil, leaky, flo_p);
    };

    // flow_estimator (all dil=1, LeakyReLU)
    conv(act0, 224, 224, 0, bufA, 128, 0, 1, 1, nullptr);   // 209->128 (act0 now dead)
    k_ctx_zero<<<7680, 256, 0, stream>>>((unsigned*)ctx0);  // zero ctx ch 34..63
    conv(bufA, 128, 128, 1, bufB, 128, 0, 1, 1, nullptr);   // 128->128
    conv(bufB, 128, 128, 2, bufA,  96, 0, 1, 1, nullptr);   // 128->96
    conv(bufA,  96,  96, 3, bufB,  64, 0, 1, 1, nullptr);   // 96->64
    conv(bufB,  64,  64, 4, ctx0,  64, 0, 1, 1, nullptr);   // 64->32 -> ctx ch 0..31
    conv(ctx0,  64,  32, 5, ctx0,  64, 32, 1, 0, flo);      // 32->2 head -> ctx ch 32..33 + f32
    // ContextNetwork (dils 1,2,4,8,16,1,1, all LeakyReLU)
    conv(ctx0,  64,  64, 6, bufA, 128, 0,  1, 1, nullptr);  // 34->128
    conv(bufA, 128, 128, 7, bufB, 128, 0,  2, 1, nullptr);  // 128->128 d2
    conv(bufB, 128, 128, 8, bufA, 128, 0,  4, 1, nullptr);  // 128->128 d4
    conv(bufA, 128, 128, 9, bufB,  96, 0,  8, 1, nullptr);  // 128->96  d8
    conv(bufB,  96,  96,10, bufA,  64, 0, 16, 1, nullptr);  // 96->64   d16
    conv(bufA,  64,  64,11, bufB,  32, 0,  1, 1, nullptr);  // 64->32
    conv(bufB,  32,  32,12, ypad,  16, 0,  1, 1, nullptr);  // 32->2

    k_final_add<<<1024, 256, 0, stream>>>(flo, ypad, (float*)d_out);
}
